// Attention_60498909331813
// MI455X (gfx1250) — matrix operand order repaired and checked by reading, not compiled, not run on hardware
//
#include <hip/hip_runtime.h>
#include <stdint.h>

typedef __attribute__((ext_vector_type(16))) _Float16 v16h;
typedef __attribute__((ext_vector_type(8)))  float    v8f;

#define B_  4
#define N_  2048
#define M_  2048
#define H_  8
#define D_  64
#define SCALE 0.125f   // 1/sqrt(64)
#define HD  (H_ * D_)  // stride between consecutive n (or m) rows = 512 floats

// LDS layout (float indices into dynamic smem):
//   Ssh    : [0      , 32768)   16 x 2048 score/prob buffer
//   red    : [32768  , 34816)   16 x 128 reduction scratch
//   rowmax : [34816  , 34832)
//   rowsum : [34832  , 34848)
//   vbuf   : [34848  , 38944)   4 waves x 2 bufs x (32 keys x 16 feat) fp32
#define SSH_OFF   0
#define RED_OFF   32768
#define RMAX_OFF  34816
#define RSUM_OFF  34832
#define VBUF_OFF  34848
#define LDS_FLOATS 38944

// Workgroup = 4 waves (128 threads), handles one (b, h, 16-row tile of N).
extern "C" __global__ __launch_bounds__(128)
void attn_fwd_wmma_async(const float* __restrict__ Q,
                         const float* __restrict__ K,
                         const float* __restrict__ V,
                         float* __restrict__ outV,   // [B,N,H,D]
                         float* __restrict__ outA) { // [B,H,N,M]
    extern __shared__ float smem[];
    float* Ssh    = smem + SSH_OFF;
    float* red    = smem + RED_OFF;
    float* rowmax = smem + RMAX_OFF;
    float* rowsum = smem + RSUM_OFF;

    const int tid  = threadIdx.x;
    const int wave = tid >> 5;
    const int lane = tid & 31;
    const int lrow = lane & 15;   // row (A frag) / col (B,C frags)
    const int lhi  = lane >> 4;   // half-wave selector

    const int blk  = blockIdx.x;
    const int nblk = blk & (N_ / 16 - 1);
    const int h    = (blk / (N_ / 16)) & (H_ - 1);
    const int b    = blk / ((N_ / 16) * H_);
    const int n0   = nblk * 16;

    // base offsets: X[((b*N + n)*H + h)*D + d]
    const size_t qbase  = (((size_t)b * N_ + n0) * H_ + h) * D_;
    const size_t kvbase = ((size_t)b * M_ * H_ + h) * D_;   // + m*HD

    // ---- Load shared Q tile as two f16 A-fragments (features 0..31, 32..63) ----
    v16h aq0, aq1;
    {
        const float* qrow = Q + qbase + (size_t)lrow * HD;
        const int d0 = lhi * 16;
        #pragma unroll
        for (int f = 0; f < 2; ++f) {
            const float4* pA = (const float4*)(qrow + f * 32 + d0);
            const float4* pB = (const float4*)(qrow + f * 32 + d0 + 8);
            float4 x0 = pA[0], x1 = pA[1];
            float4 y0 = pB[0], y1 = pB[1];
            v16h a;
            a[0]=(_Float16)x0.x; a[1]=(_Float16)x0.y; a[2]=(_Float16)x0.z; a[3]=(_Float16)x0.w;
            a[4]=(_Float16)x1.x; a[5]=(_Float16)x1.y; a[6]=(_Float16)x1.z; a[7]=(_Float16)x1.w;
            a[8]=(_Float16)y0.x; a[9]=(_Float16)y0.y; a[10]=(_Float16)y0.z; a[11]=(_Float16)y0.w;
            a[12]=(_Float16)y1.x; a[13]=(_Float16)y1.y; a[14]=(_Float16)y1.z; a[15]=(_Float16)y1.w;
            if (f == 0) aq0 = a; else aq1 = a;
        }
    }

    // ---- S = scale * Q K^T, waves split the M tiles ----
    for (int mt = wave; mt < M_ / 16; mt += 4) {
        const float* krow = K + kvbase + (size_t)(mt * 16 + lrow) * HD;
        const int dk = lhi * 16;
        v16h bk0, bk1;
        {
            const float4* p0 = (const float4*)(krow + dk);
            const float4* p1 = (const float4*)(krow + 32 + dk);
            float4 x0 = p0[0], x1 = p0[1], x2 = p0[2], x3 = p0[3];
            bk0[0]=(_Float16)x0.x; bk0[1]=(_Float16)x0.y; bk0[2]=(_Float16)x0.z; bk0[3]=(_Float16)x0.w;
            bk0[4]=(_Float16)x1.x; bk0[5]=(_Float16)x1.y; bk0[6]=(_Float16)x1.z; bk0[7]=(_Float16)x1.w;
            bk0[8]=(_Float16)x2.x; bk0[9]=(_Float16)x2.y; bk0[10]=(_Float16)x2.z; bk0[11]=(_Float16)x2.w;
            bk0[12]=(_Float16)x3.x; bk0[13]=(_Float16)x3.y; bk0[14]=(_Float16)x3.z; bk0[15]=(_Float16)x3.w;
            float4 y0 = p1[0], y1 = p1[1], y2 = p1[2], y3 = p1[3];
            bk1[0]=(_Float16)y0.x; bk1[1]=(_Float16)y0.y; bk1[2]=(_Float16)y0.z; bk1[3]=(_Float16)y0.w;
            bk1[4]=(_Float16)y1.x; bk1[5]=(_Float16)y1.y; bk1[6]=(_Float16)y1.z; bk1[7]=(_Float16)y1.w;
            bk1[8]=(_Float16)y2.x; bk1[9]=(_Float16)y2.y; bk1[10]=(_Float16)y2.z; bk1[11]=(_Float16)y2.w;
            bk1[12]=(_Float16)y3.x; bk1[13]=(_Float16)y3.y; bk1[14]=(_Float16)y3.z; bk1[15]=(_Float16)y3.w;
        }
        v8f c = {};
        c = __builtin_amdgcn_wmma_f32_16x16x32_f16(false, aq0, false, bk0, (short)0, c, false, false);
        c = __builtin_amdgcn_wmma_f32_16x16x32_f16(false, aq1, false, bk1, (short)0, c, false, false);
        #pragma unroll
        for (int r = 0; r < 8; ++r) {
            Ssh[(r + lhi * 8) * M_ + mt * 16 + lrow] = c[r] * SCALE;
        }
    }
    __syncthreads();

    // ---- Row softmax over M (16 rows x 2048 cols) ----
    for (int r = 0; r < 16; ++r) {
        float m = -1e30f;
        for (int col = tid; col < M_; col += 128) m = fmaxf(m, Ssh[r * M_ + col]);
        red[r * 128 + tid] = m;
    }
    __syncthreads();
    if (tid < 16) {
        float m = -1e30f;
        for (int i = 0; i < 128; ++i) m = fmaxf(m, red[tid * 128 + i]);
        rowmax[tid] = m;
    }
    __syncthreads();
    for (int r = 0; r < 16; ++r) {
        const float rm = rowmax[r];
        float s = 0.f;
        for (int col = tid; col < M_; col += 128) {
            float e = __expf(Ssh[r * M_ + col] - rm);
            Ssh[r * M_ + col] = e;
            s += e;
        }
        red[r * 128 + tid] = s;
    }
    __syncthreads();
    if (tid < 16) {
        float s = 0.f;
        for (int i = 0; i < 128; ++i) s += red[tid * 128 + i];
        rowsum[tid] = 1.0f / s;
    }
    __syncthreads();

    // ---- Normalize in LDS + coalesced write of attention_weight [B,H,N,M] ----
    {
        float* abase = outA + (((size_t)b * H_ + h) * N_ + n0) * M_;
        for (int idx = tid; idx < 16 * M_; idx += 128) {
            const int r = idx >> 11;          // M_ == 2048
            const float p = Ssh[idx] * rowsum[r];
            Ssh[idx] = p;
            abase[(size_t)r * M_ + (idx & (M_ - 1))] = p;
        }
    }
    __syncthreads();

    // ---- O = A * V with wave-private async-staged V tiles ----
    // Each wave owns 16 output features; per 32-key chunk it stages its
    // 32x16 fp32 V sub-tile into LDS via global_load_async_to_lds_b128
    // (4 issues/chunk, 512B each), double-buffered on ASYNCcnt.
    {
        const int dcol = wave * 16 + lrow;            // output feature column
        const int vb_wave = VBUF_OFF + wave * 1024;   // float index of this wave's buffers

        // async issue for one chunk into buffer `buf` (0/1)
        const float* vgbase = V + kvbase + wave * 16; // + m*HD + d4*4
        auto issue_chunk = [&](int mt, int buf) {
            const int fbase = vb_wave + buf * 512;    // float index
            #pragma unroll
            for (int i = 0; i < 4; ++i) {
                const int s  = i * 32 + lane;         // float4 slot 0..127
                const int m  = s >> 2;                // key row 0..31
                const int d4 = s & 3;                 // float4 within 16 features
                const uint64_t gaddr =
                    (uint64_t)(uintptr_t)(vgbase + (size_t)(mt * 32 + m) * HD + d4 * 4);
                const uint32_t laddr =
                    (uint32_t)(uintptr_t)(smem + fbase + s * 4);
                asm volatile("global_load_async_to_lds_b128 %0, %1, off"
                             :: "v"(laddr), "v"(gaddr) : "memory");
            }
        };

        v8f o = {};
        issue_chunk(0, 0);
        for (int mt = 0; mt < M_ / 32; ++mt) {
            const int buf = mt & 1;
            if (mt + 1 < M_ / 32) {
                issue_chunk(mt + 1, buf ^ 1);
                // next batch (4) may be outstanding; current batch done at <=4
                asm volatile("s_wait_asynccnt 0x4" ::: "memory");
            } else {
                asm volatile("s_wait_asynccnt 0x0" ::: "memory");
            }

            // A fragment from LDS probabilities (rows n, K = key chunk of 32)
            v16h ap;
            {
                const int kb = lhi * 16;
                const float* prow = Ssh + lrow * M_ + mt * 32;
                const float4* p0 = (const float4*)(prow + kb);
                const float4* p1 = (const float4*)(prow + kb + 8);
                float4 x0 = p0[0], x1 = p0[1], y0 = p1[0], y1 = p1[1];
                ap[0]=(_Float16)x0.x; ap[1]=(_Float16)x0.y; ap[2]=(_Float16)x0.z; ap[3]=(_Float16)x0.w;
                ap[4]=(_Float16)x1.x; ap[5]=(_Float16)x1.y; ap[6]=(_Float16)x1.z; ap[7]=(_Float16)x1.w;
                ap[8]=(_Float16)y0.x; ap[9]=(_Float16)y0.y; ap[10]=(_Float16)y0.z; ap[11]=(_Float16)y0.w;
                ap[12]=(_Float16)y1.x; ap[13]=(_Float16)y1.y; ap[14]=(_Float16)y1.z; ap[15]=(_Float16)y1.w;
            }
            // V fragment from the staged LDS tile: vbuf[m][feat], feat = lrow
            v16h bv;
            {
                const int kvb = lhi * 16;
                const float* vt = smem + vb_wave + buf * 512 + lrow;
                #pragma unroll
                for (int e = 0; e < 16; ++e)
                    bv[e] = (_Float16)vt[(kvb + e) * 16];
            }
            o = __builtin_amdgcn_wmma_f32_16x16x32_f16(false, ap, false, bv, (short)0, o, false, false);
        }
        // store queried_values [B,N,H,D]
        #pragma unroll
        for (int r = 0; r < 8; ++r) {
            const int row = r + lhi * 8;
            outV[(((size_t)b * N_ + n0 + row) * H_ + h) * D_ + dcol] = o[r];
        }
    }
}

extern "C" void kernel_launch(void* const* d_in, const int* in_sizes, int n_in,
                              void* d_out, int out_size, void* d_ws, size_t ws_size,
                              hipStream_t stream) {
    (void)in_sizes; (void)n_in; (void)out_size; (void)d_ws; (void)ws_size;
    const float* Q = (const float*)d_in[0];
    const float* K = (const float*)d_in[1];
    const float* V = (const float*)d_in[2];
    float* outV = (float*)d_out;                               // [B,N,H,D]
    float* outA = outV + (size_t)B_ * N_ * H_ * D_;            // [B,H,N,M]

    const size_t lds_bytes = (size_t)LDS_FLOATS * sizeof(float);
    dim3 grid(B_ * H_ * (N_ / 16));
    dim3 block(128);
    attn_fwd_wmma_async<<<grid, block, lds_bytes, stream>>>(Q, K, V, outV, outA);
}
